// AdditiveAttention_59897613910645
// MI455X (gfx1250) — matrix operand order repaired and checked by reading, not compiled, not run on hardware
//
#include <hip/hip_runtime.h>
#include <hip/hip_bf16.h>
#include <math.h>
#include <stdint.h>

// Problem constants (match reference)
#define B_    16
#define Q_    128
#define K_    1024
#define QDIM_ 256
#define H_    64
#define VDIM_ 256
#define KT    128        // k-tile for fused attention
#define NEGV  (-1.0e6f)

typedef __bf16 bf16_t;
typedef __attribute__((ext_vector_type(16))) __bf16 v16bf;
typedef __attribute__((ext_vector_type(8)))  __bf16 v8bf;
typedef __attribute__((ext_vector_type(8)))  float  v8f;
typedef __attribute__((ext_vector_type(4)))  uint32_t u32x4;
typedef __attribute__((ext_vector_type(4)))  int      i32x4;
typedef __attribute__((ext_vector_type(8)))  int      i32x8;

#if defined(__has_builtin)
#if __has_builtin(__builtin_amdgcn_tensor_load_to_lds) && \
    __has_builtin(__builtin_amdgcn_s_wait_tensorcnt)
#define HAVE_TDM 1
#warning CDNA5_PROBE: tensor_load_to_lds + s_wait_tensorcnt AVAILABLE (TDM path enabled)
#else
#warning CDNA5_PROBE: TDM builtins NOT available (float4 fallback path)
#endif
#if __has_builtin(__builtin_amdgcn_tanhf)
#define HAVE_TANH 1
#warning CDNA5_PROBE: __builtin_amdgcn_tanhf AVAILABLE (v_tanh_f32 path)
#else
#warning CDNA5_PROBE: __builtin_amdgcn_tanhf NOT available (libm tanhf path)
#endif
#endif

__device__ __forceinline__ float fast_tanh(float x) {
#ifdef HAVE_TANH
    return __builtin_amdgcn_tanhf(x);
#else
    return tanhf(x);
#endif
}

// ---------------------------------------------------------------------------
// Kernel 1: Out[M,64] = X[M,256] @ W[64,256]^T  via v_wmma_f32_16x16x32_bf16
// One wave per 16x16 output tile; K loop 256/32 = 8 WMMAs.
// ---------------------------------------------------------------------------
__global__ __launch_bounds__(32)
void proj_wmma(const float* __restrict__ X, const float* __restrict__ W,
               float* __restrict__ Out) {
    const int mt   = blockIdx.x;
    const int nt   = blockIdx.y;
    const int lane = threadIdx.x & 31;
    const int row  = lane & 15;   // M row (A) / N col (B,C,D)
    const int half = lane >> 4;

    const float* xrow = X + (size_t)(mt * 16 + row) * QDIM_;
    const float* wrow = W + (size_t)(nt * 16 + row) * QDIM_;  // B[k][n] = W[n][k]

    v8f c = {};
    for (int kc = 0; kc < QDIM_; kc += 32) {
        v16bf a, b;
        // A 16x32 bf16: lane row = lane%16; two contiguous runs of 8:
        //   a[0..7]  <- K = half*8 + 0..7 ; a[8..15] <- K = 16 + half*8 + 0..7
        #pragma unroll
        for (int i = 0; i < 8; ++i) {
            a[i]     = (bf16_t)xrow[kc + half * 8 + i];
            a[i + 8] = (bf16_t)xrow[kc + 16 + half * 8 + i];
        }
        // B 32x16 bf16: lane col = lane%16; same K order as A
        #pragma unroll
        for (int i = 0; i < 8; ++i) {
            b[i]     = (bf16_t)wrow[kc + half * 8 + i];
            b[i + 8] = (bf16_t)wrow[kc + 16 + half * 8 + i];
        }
        c = __builtin_amdgcn_wmma_f32_16x16x32_bf16(false, a, false, b,
                                                    (short)0, c, false, false);
    }
    // C/D layout: VGPR r -> M = r + 8*half, N = lane%16
    #pragma unroll
    for (int r = 0; r < 8; ++r)
        Out[(size_t)(mt * 16 + r + 8 * half) * H_ + nt * 16 + row] = c[r];
}

// ---------------------------------------------------------------------------
// Kernel 2: fused additive-attention scores + masked online softmax + P@V.
// Block = 256 threads (8 waves) -> >=2 waves/SIMD so WMMA (TRANS-tracked)
// co-executes with the tanh/exp VALU stream of the SIMD-mate wave.
// One (batch, 16-query tile) per block; kp k-tile staged by the TDM.
// ---------------------------------------------------------------------------
__global__ __launch_bounds__(256)
void attn_fused(const float* __restrict__ qp, const float* __restrict__ kp,
                const float* __restrict__ values, const float* __restrict__ Wv,
                const int* __restrict__ valid_lens, float* __restrict__ out) {
    __shared__ float  qp_s[16][64];          // 4 KB
    __shared__ float  wv_s[64];              // 256 B
    __shared__ float  kp_s[KT][65];          // 33.3 KB (stride 65: conflict-free)
    __shared__ float  s_s[16][128];          // 8 KB
    __shared__ bf16_t p_s[16][136];          // 4.25 KB (272B rows: 16B-aligned runs)
    __shared__ float  m_s[16], l_s[16], scale_s[16];

    const int tid   = threadIdx.x;
    const int lane  = tid & 31;
    const int wave  = tid >> 5;              // 0..7
    const int row16 = lane & 15;
    const int half  = lane >> 4;

    const int b   = blockIdx.x >> 3;         // Q_/16 = 8 q-tiles per batch
    const int qt0 = (blockIdx.x & 7) * 16;

    // stage query projections + Wv
    for (int i = tid; i < 16 * 64; i += 256)
        qp_s[i >> 6][i & 63] = qp[(size_t)(b * Q_ + qt0) * H_ + i];
    if (tid < 64) wv_s[tid] = Wv[tid];
    if (tid < 16) { m_s[tid] = -3.0e38f; l_s[tid] = 0.0f; }

    const int vlen   = valid_lens[b];
    const int ntiles = (vlen + KT - 1) / KT;   // tiles with kt0 >= vlen contribute 0

    v8f acc[2] = {};   // wave owns N-tiles ct = wave*2 + j (VDIM = 16 tiles of 16)

    __syncthreads();

    for (int t = 0; t < ntiles; ++t) {
        const int kt0 = t * KT;
        const float* ksrc = kp + (size_t)(b * K_ + kt0) * H_;

        // ---- 1. stage kp tile: TDM (tensor_load_to_lds) or float4 fallback ----
#ifdef HAVE_TDM
        if (wave == 0) {
            // D# group 0: count=1 | lds_addr | 57-bit global addr | type=2
            const uint64_t ga = (uint64_t)(uintptr_t)ksrc;
            u32x4 g0;
            g0[0] = 1u;                                        // count=1, user mode
            g0[1] = (uint32_t)(uintptr_t)&kp_s[0][0];          // LDS byte address
            g0[2] = (uint32_t)ga;                              // global_addr[31:0]
            g0[3] = (uint32_t)((ga >> 32) & 0x1FFFFFFu) | (2u << 30); // [56:32]|type=2
            // D# group 1: 2D tile 64x128 of 4B elems, row stride 64,
            // pad 1 DWORD every 64 DWORDs  -> kp_s[128][65] layout.
            i32x8 g1;
            g1[0] = (2 << 16) | (1 << 20) | (5 << 22);         // data_size=4B, pad_en,
                                                               // interval=64dw, amt=1dw
            g1[1] = (H_ & 0xFFFF) << 16;                       // tensor_dim0 = 64
            g1[2] = (KT & 0xFFFF) << 16;                       // tensor_dim1 = 128
            g1[3] = (H_ & 0xFFFF) << 16;                       // tile_dim0 = 64
            g1[4] = KT;                                        // tile_dim1 = 128
            g1[5] = H_;                                        // dim0_stride = 64
            g1[6] = (H_ & 0xFFFF) << 16;                       // dim1_stride lo (unused 2D)
            g1[7] = 0;
            i32x4 gz = {0, 0, 0, 0};
#if __clang_major__ >= 23
            i32x8 gz8 = {0, 0, 0, 0, 0, 0, 0, 0};
            __builtin_amdgcn_tensor_load_to_lds(g0, g1, gz, gz, gz8, 0);
#else
            __builtin_amdgcn_tensor_load_to_lds(g0, g1, gz, gz, 0);
#endif
            __builtin_amdgcn_s_wait_tensorcnt(0);
        }
#else
        {
            const float4* src = (const float4*)ksrc;
            for (int f = tid; f < KT * 64 / 4; f += 256) {
                float4 v = src[f];
                const int r  = f >> 4;
                const int c0 = (f & 15) * 4;
                kp_s[r][c0 + 0] = v.x; kp_s[r][c0 + 1] = v.y;
                kp_s[r][c0 + 2] = v.z; kp_s[r][c0 + 3] = v.w;
            }
        }
#endif
        if (tid < 128 && t + 1 < ntiles) {   // global_prefetch_b8 for next tile
            const char* nxt = (const char*)(ksrc + (size_t)KT * H_) + tid * 256;
            __builtin_prefetch(nxt, 0, 1);
        }
        __syncthreads();

        // ---- 2. scores: thread computes s[qb+q][k], k = tid&127, 8 rows ----
        {
            const int k  = tid & 127;
            const int qb = (tid >> 7) * 8;     // rows 0..7 or 8..15
            float pq[8];
            #pragma unroll
            for (int q = 0; q < 8; ++q) pq[q] = 0.0f;
            const bool masked_me = (kt0 + k) >= vlen;
            #pragma unroll
            for (int hb = 0; hb < 4; ++hb) {
                float kh[16], wvb[16];
                #pragma unroll
                for (int j = 0; j < 16; ++j) {
                    kh[j]  = kp_s[k][hb * 16 + j];
                    wvb[j] = wv_s[hb * 16 + j];
                }
                #pragma unroll
                for (int q = 0; q < 8; ++q) {
                    float a = 0.0f;
                    #pragma unroll
                    for (int j = 0; j < 16; ++j)
                        a += wvb[j] * fast_tanh(qp_s[qb + q][hb * 16 + j] + kh[j]);
                    pq[q] += a;
                }
            }
            #pragma unroll
            for (int q = 0; q < 8; ++q)
                s_s[qb + q][k] = masked_me ? NEGV : pq[q];
        }
        __syncthreads();

        // ---- 3. online-softmax row stats; wave owns rows 2*wave, 2*wave+1 ----
        #pragma unroll
        for (int qr = 0; qr < 2; ++qr) {
            const int q = wave * 2 + qr;
            float v0 = s_s[q][lane],      v1 = s_s[q][lane + 32],
                  v2 = s_s[q][lane + 64], v3 = s_s[q][lane + 96];
            float mx = fmaxf(fmaxf(v0, v1), fmaxf(v2, v3));
            #pragma unroll
            for (int off = 16; off >= 1; off >>= 1)
                mx = fmaxf(mx, __shfl_xor(mx, off, 32));
            const float m_old = m_s[q];
            const float m_new = fmaxf(m_old, mx);
            float e0 = __expf(v0 - m_new), e1 = __expf(v1 - m_new),
                  e2 = __expf(v2 - m_new), e3 = __expf(v3 - m_new);
            p_s[q][lane]      = (bf16_t)e0; p_s[q][lane + 32] = (bf16_t)e1;
            p_s[q][lane + 64] = (bf16_t)e2; p_s[q][lane + 96] = (bf16_t)e3;
            float sum = e0 + e1 + e2 + e3;
            #pragma unroll
            for (int off = 16; off >= 1; off >>= 1)
                sum += __shfl_xor(sum, off, 32);
            if (lane == 0) {
                const float sc = __expf(m_old - m_new);
                l_s[q]     = l_s[q] * sc + sum;
                m_s[q]     = m_new;
                scale_s[q] = sc;
            }
        }
        __syncthreads();

        // ---- 4. rescale acc, then acc += P(16xKT) @ V(KTx256) via WMMA ----
        {
            float sc[8];
            #pragma unroll
            for (int r = 0; r < 8; ++r) sc[r] = scale_s[r + 8 * half];
            #pragma unroll
            for (int j = 0; j < 2; ++j)
                #pragma unroll
                for (int r = 0; r < 8; ++r) acc[j][r] *= sc[r];

            #pragma unroll
            for (int ks = 0; ks < 4; ++ks) {       // KT=128 -> 4 x K32 steps
                // A operand: two 16B-aligned 8-half runs -> 2x ds_load_b128
                v8bf lo = *(const v8bf*)&p_s[row16][ks * 32 + half * 8];
                v8bf hi = *(const v8bf*)&p_s[row16][ks * 32 + 16 + half * 8];
                v16bf a;
                #pragma unroll
                for (int i = 0; i < 8; ++i) { a[i] = lo[i]; a[i + 8] = hi[i]; }
                #pragma unroll
                for (int j = 0; j < 2; ++j) {
                    const int ct = wave * 2 + j;
                    const float* vsrc = values
                        + (size_t)(b * K_ + kt0 + ks * 32 + half * 16) * VDIM_
                        + ct * 16 + row16;
                    v16bf bb;
                    #pragma unroll
                    for (int i = 0; i < 16; ++i)
                        bb[i] = (bf16_t)vsrc[(ptrdiff_t)(i < 8 ? i - half * 8 : i + 8 - half * 8) * VDIM_];
                    acc[j] = __builtin_amdgcn_wmma_f32_16x16x32_bf16(
                        false, a, false, bb, (short)0, acc[j], false, false);
                }
            }
        }
        __syncthreads();
    }

    // ---- epilogue: out = acc / l ----
    float linv[8];
    #pragma unroll
    for (int r = 0; r < 8; ++r) linv[r] = 1.0f / l_s[r + 8 * half];
    #pragma unroll
    for (int j = 0; j < 2; ++j) {
        const int ct = wave * 2 + j;
        #pragma unroll
        for (int r = 0; r < 8; ++r)
            out[(size_t)(b * Q_ + qt0 + r + 8 * half) * VDIM_ + ct * 16 + row16] =
                acc[j][r] * linv[r];
    }
}

// ---------------------------------------------------------------------------
// Launcher
// ---------------------------------------------------------------------------
extern "C" void kernel_launch(void* const* d_in, const int* in_sizes, int n_in,
                              void* d_out, int out_size, void* d_ws, size_t ws_size,
                              hipStream_t stream) {
    (void)in_sizes; (void)n_in; (void)out_size; (void)ws_size;
    const float* queries = (const float*)d_in[0];
    const float* keys    = (const float*)d_in[1];
    const float* values  = (const float*)d_in[2];
    const int*   vlens   = (const int*)  d_in[3];
    const float* Wq      = (const float*)d_in[4];
    const float* Wk      = (const float*)d_in[5];
    const float* Wv      = (const float*)d_in[6];
    float* out = (float*)d_out;

    // workspace: kp [B*K, 64] then qp [B*Q, 64]  (4.5 MB total, fp32)
    float* kp = (float*)d_ws;
    float* qp = kp + (size_t)B_ * K_ * H_;

    proj_wmma<<<dim3(B_ * K_ / 16, H_ / 16), 32, 0, stream>>>(keys,    Wk, kp);
    proj_wmma<<<dim3(B_ * Q_ / 16, H_ / 16), 32, 0, stream>>>(queries, Wq, qp);
    attn_fused<<<dim3(B_ * Q_ / 16), 256, 0, stream>>>(qp, kp, values, Wv, vlens, out);
}
